// CustomModel_85633057948283
// MI455X (gfx1250) — matrix operand order repaired and checked by reading, not compiled, not run on hardware
//
#include <hip/hip_runtime.h>
#include <hip/hip_bf16.h>

// ---------------------------------------------------------------------------
// Top-2-of-8 MoE, 2 layers. Sparse grouped-GEMM on v_wmma_f32_16x16x32_bf16.
// Fast path: weights are pre-converted ONCE per call into a bf16,
// WMMA-fragment-tiled layout in the workspace (32x16 K*N tiles stored
// lane-major, 32B per lane) so each B fragment is 2 coalesced b128 loads
// with zero conversion VALU. Roofline: 268MB fp32 read + 134MB bf16
// write/read ~ 17-23us @ 23.3TB/s; sparse compute ~69 GFLOP bf16.
// ---------------------------------------------------------------------------

#define D_DIM 1024
#define F_DIM 2048
#define E_NUM 8
#define L_NUM 2
#define NTOK  2048           // B*S
#define CAP   2048           // per-expert list capacity

typedef __attribute__((ext_vector_type(16))) __bf16          v16bf;
typedef __attribute__((ext_vector_type(2)))  __bf16          v2bf;
typedef __attribute__((ext_vector_type(16))) unsigned short  v16u;
typedef __attribute__((ext_vector_type(8)))  unsigned int    v8u;
typedef __attribute__((ext_vector_type(8)))  float           v8f;

__device__ __forceinline__ unsigned short f2bf(float f) {
    unsigned u = __builtin_bit_cast(unsigned, f);
    u += 0x7fffu + ((u >> 16) & 1u);
    return (unsigned short)(u >> 16);
}

// pack two fp32 -> one dword of 2x bf16 (RNE)
__device__ __forceinline__ unsigned pkbf(float a, float b) {
#if __has_builtin(__builtin_amdgcn_cvt_pk_bf16_f32)
    return __builtin_bit_cast(unsigned, __builtin_amdgcn_cvt_pk_bf16_f32(a, b));
#else
    return (unsigned)f2bf(a) | ((unsigned)f2bf(b) << 16);
#endif
}

// ---------------------------------------------------------------------------
__global__ void zero_counts_kernel(int* __restrict__ counts) {
    if (threadIdx.x < E_NUM) counts[threadIdx.x] = 0;
}

__global__ void copy_f4_kernel(const float4* __restrict__ src,
                               float4* __restrict__ dst, int n) {
    int i = blockIdx.x * blockDim.x + threadIdx.x;
    if (i < n) dst[i] = src[i];
}

// ---------------------------------------------------------------------------
// Convert fp32 weights [EL][K][N] -> bf16 tiles in WMMA-B fragment order.
// Tile (kt,nt) of one "expert" block: 32 lanes x 16 bf16; lane=hi*16+lo holds
// elements i=0..15 with K = kt*32 + 16*hi + i, N = nt*16 + lo.
// Per-expert tile order: nt-major, kt contiguous (streams in the K loop).
__global__ void convert_tiles_kernel(const float* __restrict__ src,
                                     unsigned short* __restrict__ dst,
                                     int K, int N, int total) {
    int g = blockIdx.x * blockDim.x + threadIdx.x;
    if (g >= total) return;
    const int lane = g & 31;
    int rest = g >> 5;
    const int KT = K >> 5;                  // tiles along K
    const int kt = rest % KT;  rest /= KT;
    const int nt = rest % (N >> 4);
    const int e  = rest / (N >> 4);

    const float* s = src + ((size_t)e * K + (size_t)kt * 32 + ((lane >> 4) << 4)) * N
                         + (size_t)nt * 16 + (lane & 15);
    v8u o;
    #pragma unroll
    for (int w = 0; w < 8; ++w)
        o[w] = pkbf(s[(size_t)(2 * w) * N], s[(size_t)(2 * w + 1) * N]);
    *(v8u*)(dst + (size_t)g * 16) = o;      // 32B per thread, fully coalesced
}

// ---------------------------------------------------------------------------
// One wave32 per token: scores = |x . protos_e|, top-2, softmax over the two.
__global__ void route_kernel(const float* __restrict__ x,
                             const float* __restrict__ protos,
                             int* __restrict__ counts,
                             int* __restrict__ entries,
                             float* __restrict__ wts) {
    const int wv   = threadIdx.x >> 5;
    const int lane = threadIdx.x & 31;
    const int tok  = blockIdx.x * 8 + wv;
    if (tok >= NTOK) return;
    const float* xr = x + (size_t)tok * D_DIM;

    float s[E_NUM];
    #pragma unroll
    for (int e = 0; e < E_NUM; ++e) {
        const float* pr = protos + (size_t)e * D_DIM;
        float p = 0.f;
        for (int d = lane; d < D_DIM; d += 32) p += xr[d] * pr[d];
        #pragma unroll
        for (int off = 16; off; off >>= 1) p += __shfl_xor(p, off, 32);
        s[e] = fabsf(p);
    }
    int e1 = 0; float s1 = s[0];
    #pragma unroll
    for (int e = 1; e < E_NUM; ++e) if (s[e] > s1) { s1 = s[e]; e1 = e; }
    int e2 = -1; float s2 = -1.f;
    #pragma unroll
    for (int e = 0; e < E_NUM; ++e)
        if (e != e1 && s[e] > s2) { s2 = s[e]; e2 = e; }
    const float t  = __expf(s2 - s1);
    const float w1 = 1.f / (1.f + t);
    const float w2 = t * w1;

    if (lane == 0) {
        int slot = atomicAdd(&counts[e1], 1);
        entries[e1 * CAP + slot] = tok;  wts[e1 * CAP + slot] = w1;
        slot = atomicAdd(&counts[e2], 1);
        entries[e2 * CAP + slot] = tok;  wts[e2 * CAP + slot] = w2;
    }
}

// ---------------------------------------------------------------------------
// LDS strides: row stride == 5 (mod 64) dwords -> 32 conflict-free banks for
// the A-fragment fetch pattern (pad 10 shorts per row).
#define XA_STRIDE (D_DIM + 10)
#define H_STRIDE  (F_DIM + 10)

// Shared prologue: stage meta + X tile (bf16) into LDS. Returns false if the
// whole block should exit.
__device__ __forceinline__ bool ffn_prologue(const float* __restrict__ xin,
                                             const int* __restrict__ counts,
                                             const int* __restrict__ entries,
                                             const float* __restrict__ wts,
                                             int e, int tile,
                                             unsigned short* sX, int* sTok, float* sW) {
    const int cnt = counts[e];
    if (tile * 16 >= cnt) return false;
    const int tid = threadIdx.x;
    if (tid < 16) {
        int idx = tile * 16 + tid;
        if (idx < cnt) { sTok[tid] = entries[e * CAP + idx]; sW[tid] = wts[e * CAP + idx]; }
        else           { sTok[tid] = entries[e * CAP + cnt - 1]; sW[tid] = 0.f; }
    }
    __syncthreads();
    {
        const int row = tid >> 4;
        const float* xr = xin + (size_t)sTok[row] * D_DIM;
        for (int c = (tid & 15) * 2; c < D_DIM; c += 32)
            *(unsigned*)&sX[row * XA_STRIDE + c] = pkbf(xr[c], xr[c + 1]);
    }
    __syncthreads();
    return true;
}

// A fragment (16x32 bf16) from LDS, contiguous k order: element i is k = 16*hi + i.
__device__ __forceinline__ v16bf load_A(const unsigned short* base) {
    v16u au;
    #pragma unroll
    for (int i = 0; i < 8; ++i) { au[i] = base[i]; au[8 + i] = base[8 + i]; }
    return __builtin_bit_cast(v16bf, au);
}

// B fragment from pre-tiled bf16 weights: 32 contiguous bytes per lane.
__device__ __forceinline__ v16bf load_Bt(const unsigned short* p) {
    return __builtin_bit_cast(v16bf, *(const v8u*)p);
}

// ------------------------ fast path: pre-tiled bf16 weights -----------------
__global__ void ffn_bf16w_kernel(const float* __restrict__ xin,
                                 const unsigned short* __restrict__ W1t, // per-layer tiled
                                 const unsigned short* __restrict__ W2t,
                                 const int*   __restrict__ counts,
                                 const int*   __restrict__ entries,
                                 const float* __restrict__ wts,
                                 float* __restrict__ xout) {
    __shared__ unsigned short sX[16 * XA_STRIDE];
    __shared__ unsigned short sH[16 * H_STRIDE];
    __shared__ int   sTok[16];
    __shared__ float sW[16];

    const int e = blockIdx.y, tile = blockIdx.x;
    if (!ffn_prologue(xin, counts, entries, wts, e, tile, sX, sTok, sW)) return;

    const int tid  = threadIdx.x;
    const int wv   = tid >> 5;
    const int lane = tid & 31;
    const int hi   = lane >> 4, lo = lane & 15;

    // ---- GEMM1: h = relu(X @ W1[e]) ----
    const unsigned short* W1e = W1t + (size_t)e * D_DIM * F_DIM;
    for (int fg = 0; fg < 4; ++fg) {
        v8f acc[4] = {};
        const int ftBase = wv * 16 + fg * 4;
        for (int ks = 0; ks < D_DIM / 32; ++ks) {
            const v16bf a = load_A(&sX[lo * XA_STRIDE + ks * 32 + 16 * hi]);
            #pragma unroll
            for (int j = 0; j < 4; ++j) {
                const unsigned short* bp =
                    W1e + (((size_t)(ftBase + j) * (D_DIM / 32) + ks) * 512) + lane * 16;
                acc[j] = __builtin_amdgcn_wmma_f32_16x16x32_bf16(
                            false, a, false, load_Bt(bp), (short)0, acc[j], false, false);
            }
        }
        #pragma unroll
        for (int j = 0; j < 4; ++j) {
            const int fcol = (ftBase + j) * 16 + lo;
            #pragma unroll
            for (int r = 0; r < 8; ++r) {
                float v = acc[j][r];
                sH[(r + 8 * hi) * H_STRIDE + fcol] = f2bf(v > 0.f ? v : 0.f);
            }
        }
    }
    __syncthreads();

    // ---- GEMM2: y = (h @ W2[e]) * w ----
    const unsigned short* W2e = W2t + (size_t)e * F_DIM * D_DIM;
    for (int dg = 0; dg < 2; ++dg) {
        v8f acc[4] = {};
        const int dtBase = wv * 8 + dg * 4;
        for (int ks = 0; ks < F_DIM / 32; ++ks) {
            const v16bf a = load_A(&sH[lo * H_STRIDE + ks * 32 + 16 * hi]);
            #pragma unroll
            for (int j = 0; j < 4; ++j) {
                const unsigned short* bp =
                    W2e + (((size_t)(dtBase + j) * (F_DIM / 32) + ks) * 512) + lane * 16;
                acc[j] = __builtin_amdgcn_wmma_f32_16x16x32_bf16(
                            false, a, false, load_Bt(bp), (short)0, acc[j], false, false);
            }
        }
        #pragma unroll
        for (int j = 0; j < 4; ++j) {
            const int dc = (dtBase + j) * 16 + lo;
            #pragma unroll
            for (int r = 0; r < 8; ++r) {
                const int m = r + 8 * hi;
                atomicAdd(&xout[(size_t)sTok[m] * D_DIM + dc], acc[j][r] * sW[m]);
            }
        }
    }
}

// ------------------- fallback: fp32 weights, convert in-GEMM ----------------
__device__ __forceinline__ v16bf load_Bf32(const float* bp, size_t stride) {
    v16u bu;
    #pragma unroll
    for (int i = 0; i < 16; ++i) bu[i] = f2bf(bp[(size_t)i * stride]);
    return __builtin_bit_cast(v16bf, bu);
}

__global__ void ffn_f32w_kernel(const float* __restrict__ xin,
                                const float* __restrict__ W1,
                                const float* __restrict__ W2,
                                const int*   __restrict__ counts,
                                const int*   __restrict__ entries,
                                const float* __restrict__ wts,
                                float* __restrict__ xout) {
    __shared__ unsigned short sX[16 * XA_STRIDE];
    __shared__ unsigned short sH[16 * H_STRIDE];
    __shared__ int   sTok[16];
    __shared__ float sW[16];

    const int e = blockIdx.y, tile = blockIdx.x;
    if (!ffn_prologue(xin, counts, entries, wts, e, tile, sX, sTok, sW)) return;

    const int tid  = threadIdx.x;
    const int wv   = tid >> 5;
    const int lane = tid & 31;
    const int hi   = lane >> 4, lo = lane & 15;

    const size_t W1e = (size_t)e * D_DIM * F_DIM;
    for (int fg = 0; fg < 4; ++fg) {
        v8f acc[4] = {};
        const int ftBase = wv * 16 + fg * 4;
        for (int ks = 0; ks < D_DIM; ks += 32) {
            const v16bf a = load_A(&sX[lo * XA_STRIDE + ks + 16 * hi]);
            #pragma unroll
            for (int j = 0; j < 4; ++j) {
                const float* bp = W1 + W1e + (size_t)(ks + 16 * hi) * F_DIM
                                + (ftBase + j) * 16 + lo;
                acc[j] = __builtin_amdgcn_wmma_f32_16x16x32_bf16(
                            false, a, false, load_Bf32(bp, F_DIM), (short)0, acc[j], false, false);
            }
        }
        #pragma unroll
        for (int j = 0; j < 4; ++j) {
            const int fcol = (ftBase + j) * 16 + lo;
            #pragma unroll
            for (int r = 0; r < 8; ++r) {
                float v = acc[j][r];
                sH[(r + 8 * hi) * H_STRIDE + fcol] = f2bf(v > 0.f ? v : 0.f);
            }
        }
    }
    __syncthreads();

    const size_t W2e = (size_t)e * F_DIM * D_DIM;
    for (int dg = 0; dg < 2; ++dg) {
        v8f acc[4] = {};
        const int dtBase = wv * 8 + dg * 4;
        for (int ks = 0; ks < F_DIM; ks += 32) {
            const v16bf a = load_A(&sH[lo * H_STRIDE + ks + 16 * hi]);
            #pragma unroll
            for (int j = 0; j < 4; ++j) {
                const float* bp = W2 + W2e + (size_t)(ks + 16 * hi) * D_DIM
                                + (dtBase + j) * 16 + lo;
                acc[j] = __builtin_amdgcn_wmma_f32_16x16x32_bf16(
                            false, a, false, load_Bf32(bp, D_DIM), (short)0, acc[j], false, false);
            }
        }
        #pragma unroll
        for (int j = 0; j < 4; ++j) {
            const int dc = (dtBase + j) * 16 + lo;
            #pragma unroll
            for (int r = 0; r < 8; ++r) {
                const int m = r + 8 * hi;
                atomicAdd(&xout[(size_t)sTok[m] * D_DIM + dc], acc[j][r] * sW[m]);
            }
        }
    }
}

// ---------------------------------------------------------------------------
extern "C" void kernel_launch(void* const* d_in, const int* in_sizes, int n_in,
                              void* d_out, int out_size, void* d_ws, size_t ws_size,
                              hipStream_t stream) {
    const float* x      = (const float*)d_in[0];
    const float* protos = (const float*)d_in[1];   // [L][E][D]
    const float* W1     = (const float*)d_in[2];   // [L][E][D][F]
    const float* W2     = (const float*)d_in[3];   // [L][E][F][D]
    float* out = (float*)d_out;

    char*  ws      = (char*)d_ws;
    float* xA      = (float*)ws;                                 // 8 MB
    int*   counts  = (int*)(ws + (size_t)NTOK * D_DIM * 4);
    int*   entries = counts + 64;
    float* wvals   = (float*)(entries + E_NUM * CAP);

    const size_t perW   = (size_t)L_NUM * E_NUM * D_DIM * F_DIM; // elems per weight tensor
    const size_t stateB = 9u << 20;                              // xA + lists (padded)
    unsigned short* W1t = (unsigned short*)(ws + stateB);
    unsigned short* W2t = W1t + perW;
    const bool pre = ws_size >= stateB + 2 * perW * sizeof(unsigned short);

    const size_t lW  = (size_t)E_NUM * D_DIM * F_DIM;            // per-layer elems
    const int    nf4 = NTOK * D_DIM / 4;
    const dim3   ffnGrid(NTOK / 16, E_NUM);

    if (pre) {
        // fold L*E into one "expert" axis: per-expert tiled block = K*N elems
        const int totW1 = L_NUM * E_NUM * (F_DIM / 16) * (D_DIM / 32) * 32;
        const int totW2 = L_NUM * E_NUM * (D_DIM / 16) * (F_DIM / 32) * 32;
        convert_tiles_kernel<<<(totW1 + 255) / 256, 256, 0, stream>>>(W1, W1t, D_DIM, F_DIM, totW1);
        convert_tiles_kernel<<<(totW2 + 255) / 256, 256, 0, stream>>>(W2, W2t, F_DIM, D_DIM, totW2);
    }

    for (int l = 0; l < L_NUM; ++l) {
        const float* xi = (l == 0) ? x : xA;
        float*       xo = (l == 0) ? xA : out;
        zero_counts_kernel<<<1, 32, 0, stream>>>(counts);
        route_kernel<<<NTOK / 8, 256, 0, stream>>>(xi, protos + (size_t)l * E_NUM * D_DIM,
                                                   counts, entries, wvals);
        copy_f4_kernel<<<(nf4 + 255) / 256, 256, 0, stream>>>((const float4*)xi, (float4*)xo, nf4);
        if (pre)
            ffn_bf16w_kernel<<<ffnGrid, 256, 0, stream>>>(xi, W1t + l * lW, W2t + l * lW,
                                                          counts, entries, wvals, xo);
        else
            ffn_f32w_kernel<<<ffnGrid, 256, 0, stream>>>(xi, W1 + l * lW, W2 + l * lW,
                                                         counts, entries, wvals, xo);
    }
}
